// Model_24180665876772
// MI455X (gfx1250) — matrix operand order repaired and checked by reading, not compiled, not run on hardware
//
#include <hip/hip_runtime.h>
#include <math.h>

// ---------------- types ----------------
typedef __attribute__((ext_vector_type(16))) __bf16 v16bf;
typedef __attribute__((ext_vector_type(8)))  __bf16 v8bf;
typedef __attribute__((ext_vector_type(8)))  float  v8f;
typedef __attribute__((ext_vector_type(4)))  unsigned int v4u;
typedef __attribute__((ext_vector_type(8)))  int v8i;
typedef __attribute__((ext_vector_type(4)))  int v4i;

#define Bc     32
#define Lc     512
#define Nc     512
#define Dc     512
#define Hc     8
#define Ec     64
#define DFc    2048
#define PREDc  96
#define MODESc 64
#define MROWS  (Bc * Nc)   // 16384 rows for every big GEMM

#if defined(__has_builtin)
#if __has_builtin(__builtin_amdgcn_tensor_load_to_lds) && \
    __has_builtin(__builtin_amdgcn_s_wait_tensorcnt)
#define USE_TDM 1
#else
#define USE_TDM 0
#endif
#else
#define USE_TDM 0
#endif

static __device__ __forceinline__ v16bf mk16(v8bf lo, v8bf hi) {
  v16bf r;
#pragma unroll
  for (int i = 0; i < 8; ++i) { r[i] = lo[i]; r[i + 8] = hi[i]; }
  return r;
}

#if USE_TDM
// ---- Tensor Data Mover: 2D bf16 tile (tile_dim0=32 elems, tile_dim1=128 rows)
// global (row stride = stride0 elems) -> LDS rows padded to 40 halves (80B):
// pad_enable=1, pad_interval=3 (every 16 DWORDs), pad_amount=3 (+4 DWORDs).
// Rows >= tensor_d1 zero-fill (OOB reads return zero) -> free edge handling.
// 6-arg builtin (clang-23 / therock lane): extra int32x8 group, zero-filled.
static __device__ __forceinline__ void tdm_load_tile(unsigned lds_off,
                                                     const __bf16* gptr,
                                                     unsigned tensor_d0,
                                                     unsigned tensor_d1,
                                                     unsigned stride0) {
  unsigned long long ga = (unsigned long long)(size_t)gptr;
  v4u g0;
  g0[0] = 1u;                                      // count=1 (valid), user mode
  g0[1] = lds_off;                                 // LDS byte address
  g0[2] = (unsigned)(ga & 0xFFFFFFFFu);            // global_addr[31:0]
  g0[3] = (unsigned)((ga >> 32) & 0x1FFFFFFu)      // global_addr[56:32]
        | (2u << 30);                              // type=2 ("image")
  v8i g1;
  g1[0] = (1 << 16)                                // data_size=1 -> 2 bytes
        | (1 << 20)                                // pad_enable
        | (3 << 22)                                // pad_interval: 16 DWORDs
        | (3 << 25);                               // pad_amount:  4 DWORDs
  g1[1] = (int)((tensor_d0 & 0xFFFFu) << 16);      // [63:48] tensor_dim0 lo16
  g1[2] = (int)((tensor_d0 >> 16) | ((tensor_d1 & 0xFFFFu) << 16));
  g1[3] = (int)((tensor_d1 >> 16) | (32u << 16));  // [127:112] tile_dim0 = 32
  g1[4] = 128;                                     // tile_dim1=128, tile_dim2=0
  g1[5] = (int)stride0;                            // tensor_dim0_stride lo32
  g1[6] = 0;                                       // stride0 hi, stride1 lo
  g1[7] = 0;
  v4i z4; z4[0] = 0; z4[1] = 0; z4[2] = 0; z4[3] = 0;  // 2D: groups 2/3 unused
  v8i z8;
#pragma unroll
  for (int q = 0; q < 8; ++q) z8[q] = 0;
  __builtin_amdgcn_tensor_load_to_lds(g0, g1, z4, z4, z8, 0);
}
#endif

// ---------------- bf16 WMMA GEMM: out = act(A[M,K] * Bt[N,K]^T + bias + resid)
// A row-major bf16 (lda=K), Bt row-major bf16 [N][K] (weights pre-transposed).
// Block tile 128x128, 8 waves each 32x64 = 2x4 v_wmma_f32_16x16x32_bf16.
// Staging: TDM double-buffered (DMA of tile t+1 overlaps WMMA of tile t),
// one workgroup barrier per 32-deep K step.
__global__ __launch_bounds__(256)
void k_gemm_bf16(const __bf16* __restrict__ A, const __bf16* __restrict__ Bt,
                 int M, int N, int K,
                 const float* __restrict__ bias, const float* __restrict__ resid,
                 int act, float* __restrict__ outF, __bf16* __restrict__ outB) {
  __shared__ __bf16 As[2][128 * 40];   // rows stride 40 halves = TDM pad layout
  __shared__ __bf16 Bs[2][128 * 40];

  const int tid   = threadIdx.x;
  const int mBase = blockIdx.y * 128;
  const int nBase = blockIdx.x * 128;
  const int lane  = tid & 31;
  const int wv    = tid >> 5;
  const int wm    = (wv >> 1) << 5;  // 0,32,64,96
  const int wn    = (wv & 1) << 6;   // 0,64
  const int lh    = lane >> 4;
  const int lm    = lane & 15;

  v8f acc[2][4] = {};
  const int T = K >> 5;

#if USE_TDM
  const unsigned nrem = (unsigned)(N - nBase);
  if (wv == 0) {
    tdm_load_tile((unsigned)(size_t)&As[0][0], A + (size_t)mBase * K,
                  (unsigned)K, (unsigned)(M - mBase), (unsigned)K);
    tdm_load_tile((unsigned)(size_t)&Bs[0][0], Bt + (size_t)nBase * K,
                  (unsigned)K, nrem, (unsigned)K);
    __builtin_amdgcn_s_wait_tensorcnt(0);
  }
  __syncthreads();
#endif

  for (int t = 0; t < T; ++t) {
#if USE_TDM
    const int cur = t & 1;
    if (wv == 0 && t + 1 < T) {   // kick DMA for next tile into the other buffer
      const int k0n = (t + 1) << 5;
      tdm_load_tile((unsigned)(size_t)&As[cur ^ 1][0],
                    A + (size_t)mBase * K + k0n,
                    (unsigned)(K - k0n), (unsigned)(M - mBase), (unsigned)K);
      tdm_load_tile((unsigned)(size_t)&Bs[cur ^ 1][0],
                    Bt + (size_t)nBase * K + k0n,
                    (unsigned)(K - k0n), nrem, (unsigned)K);
    }
#else
    const int cur = 0;
    const int k0 = t << 5;
    // stage A tile: 128x32 halves = 512 chunks of 8 halves
#pragma unroll
    for (int s = 0; s < 2; ++s) {
      int c   = tid + s * 256;
      int row = c >> 2;
      int ko  = (c & 3) << 3;
      v8bf v = *(const v8bf*)(A + (size_t)(mBase + row) * K + k0 + ko);
      *(v8bf*)(&As[0][0] + row * 40 + ko) = v;
    }
    // stage B tile: 128x32 halves, zero-fill rows beyond N
#pragma unroll
    for (int s = 0; s < 2; ++s) {
      int c   = tid + s * 256;
      int row = c >> 2;
      int ko  = (c & 3) << 3;
      int gn  = nBase + row;
      v8bf v;
#pragma unroll
      for (int q = 0; q < 8; ++q) v[q] = (__bf16)0.0f;
      if (gn < N) v = *(const v8bf*)(Bt + (size_t)gn * K + k0 + ko);
      *(v8bf*)(&Bs[0][0] + row * 40 + ko) = v;
    }
    __syncthreads();
#endif

    // gather fragments per the ISA 16x32 A / 32x16 B lane layouts
    v16bf afr[2], bfr[4];
#pragma unroll
    for (int i = 0; i < 2; ++i) {
      const __bf16* ap = &As[cur][0] + (wm + i * 16 + lm) * 40;
      v8bf lo = *(const v8bf*)(ap + (lh ? 8 : 0));
      v8bf hi = *(const v8bf*)(ap + (lh ? 24 : 16));
      afr[i] = mk16(lo, hi);
    }
#pragma unroll
    for (int j = 0; j < 4; ++j) {
      const __bf16* bp = &Bs[cur][0] + (wn + j * 16 + lm) * 40;
      v8bf lo = *(const v8bf*)(bp + (lh ? 8 : 0));
      v8bf hi = *(const v8bf*)(bp + (lh ? 24 : 16));
      bfr[j] = mk16(lo, hi);
    }
#pragma unroll
    for (int i = 0; i < 2; ++i)
#pragma unroll
      for (int j = 0; j < 4; ++j)
        acc[i][j] = __builtin_amdgcn_wmma_f32_16x16x32_bf16(
            false, afr[i], false, bfr[j], (short)0, acc[i][j], false, false);

#if USE_TDM
    if (wv == 0) __builtin_amdgcn_s_wait_tensorcnt(0);
#endif
    __syncthreads();
  }

  // epilogue: C layout M = r + 8*(lane>>4), N = lane&15
#pragma unroll
  for (int i = 0; i < 2; ++i)
#pragma unroll
    for (int j = 0; j < 4; ++j)
#pragma unroll
      for (int r = 0; r < 8; ++r) {
        int gm = mBase + wm + i * 16 + r + (lh << 3);
        int gn = nBase + wn + j * 16 + lm;
        if (gn < N) {
          float v = acc[i][j][r];
          if (bias)  v += bias[gn];
          if (resid) v += resid[(size_t)gm * N + gn];
          if (act == 1) v = 0.5f * v * (1.0f + erff(v * 0.70710678118654752f));
          size_t o = (size_t)gm * N + gn;
          if (outF) outF[o] = v;
          if (outB) outB[o] = (__bf16)v;
        }
      }
}

// ---------------- series_decomp on input: x_enc[B,L,N] -> seasonalT/trendT [B,N,L] bf16
__global__ void k_decomp_in(const float* __restrict__ x,
                            __bf16* __restrict__ seasT, __bf16* __restrict__ trendT) {
  int t = blockIdx.x * blockDim.x + threadIdx.x;      // (b,n)
  if (t >= Bc * Nc) return;
  const float* xp = x + (size_t)(t >> 9) * Lc * Nc + (t & (Nc - 1));
  float sum = 0.f;
#pragma unroll
  for (int j = -12; j <= 12; ++j) sum += xp[(size_t)(j < 0 ? 0 : j) * Nc];
  __bf16* sT = seasT + (size_t)t * Lc;
  __bf16* tT = trendT + (size_t)t * Lc;
  for (int l = 0; l < Lc; ++l) {
    float mean = sum * (1.0f / 25.0f);
    float xv = xp[(size_t)l * Nc];
    tT[l] = (__bf16)mean;
    sT[l] = (__bf16)(xv - mean);
    int la = l + 13; if (la > Lc - 1) la = Lc - 1;
    int ls = l - 12; if (ls < 0) ls = 0;
    sum += xp[(size_t)la * Nc] - xp[(size_t)ls * Nc];
  }
}

// ---------------- series_decomp over token axis: x[B,N,D] -> seasonal f32 + bf16
__global__ void k_decomp_tok(const float* __restrict__ x,
                             float* __restrict__ outF, __bf16* __restrict__ outB) {
  int t = blockIdx.x * blockDim.x + threadIdx.x;      // (b,d)
  if (t >= Bc * Dc) return;
  size_t base = (size_t)(t >> 9) * Nc * Dc + (t & (Dc - 1));
  const float* xp = x + base;
  float* oF = outF + base;
  __bf16* oB = outB + base;
  float sum = 0.f;
#pragma unroll
  for (int j = -12; j <= 12; ++j) sum += xp[(size_t)(j < 0 ? 0 : j) * Dc];
  for (int n = 0; n < Nc; ++n) {
    float v = xp[(size_t)n * Dc] - sum * (1.0f / 25.0f);
    oF[(size_t)n * Dc] = v;
    oB[(size_t)n * Dc] = (__bf16)v;
    int na = n + 13; if (na > Nc - 1) na = Nc - 1;
    int ns = n - 12; if (ns < 0) ns = 0;
    sum += xp[(size_t)na * Dc] - xp[(size_t)ns * Dc];
  }
}

// ---------------- complex mode mixing: out_ft[b,h,o,m] = sum_i x_ft[b,h,i,m]*fw[h,i,o,m]
// X,Y layout [B, 512(h*64+ch), 128(Re0..63|Im0..63)], one block per (h,m).
__global__ __launch_bounds__(256)
void k_mode_mix(const float* __restrict__ X, const float* __restrict__ fwr,
                const float* __restrict__ fwi, __bf16* __restrict__ Y) {
  __shared__ float Fr[64][65];
  __shared__ float Fi[64][65];
  __shared__ float Xs[4][64][2];
  const int tid = threadIdx.x;
  const int h = blockIdx.x >> 6;
  const int m = blockIdx.x & 63;
  for (int idx = tid; idx < 4096; idx += 256) {
    int i = idx >> 6, o = idx & 63;
    size_t f = ((size_t)((h * 64 + i) * 64 + o) << 6) + m;
    Fr[i][o] = fwr[f];
    Fi[i][o] = fwi[f];
  }
  __syncthreads();
  const int o = tid & 63, bq = tid >> 6;
  for (int bp = 0; bp < 8; ++bp) {
    for (int idx = tid; idx < 512; idx += 256) {
      int bb = idx >> 7, rest = idx & 127;
      int i = rest >> 1, ri = rest & 1;
      size_t base = ((size_t)(bp * 4 + bb) * 512 + h * 64 + i) * 128;
      Xs[bb][i][ri] = X[base + (ri ? (64 + m) : m)];
    }
    __syncthreads();
    float orr = 0.f, oii = 0.f;
#pragma unroll 8
    for (int i = 0; i < 64; ++i) {
      float xr = Xs[bq][i][0], xi = Xs[bq][i][1];
      float fr = Fr[i][o], fi = Fi[i][o];
      orr += xr * fr - xi * fi;
      oii += xr * fi + xi * fr;
    }
    size_t base = ((size_t)(bp * 4 + bq) * 512 + h * 64 + o) * 128;
    Y[base + m]      = (__bf16)orr;
    Y[base + 64 + m] = (__bf16)oii;
    __syncthreads();
  }
}

// ---------------- DFT / iDFT bases (bf16, pre-transposed for the GEMM B operand)
__global__ void k_fill_basis(__bf16* __restrict__ dftT, __bf16* __restrict__ idftT) {
  int idx = blockIdx.x * blockDim.x + threadIdx.x;
  if (idx >= 128 * 512) return;
  {   // dftT[j][n], j<64: cos(2pi j n/512); j>=64: -sin  (cols 64.. = Im of rfft)
    int j = idx >> 9, n = idx & 511;
    int mm = j & 63;
    float ang = 6.283185307179586f * (float)(mm * n) * (1.0f / 512.0f);
    dftT[idx] = (__bf16)((j < 64) ? cosf(ang) : -sinf(ang));
  }
  {   // idftT[n][j]: irfft with modes>=64 zero; imag of DC ignored
    int n = idx >> 7, j = idx & 127;
    int mm = j & 63;
    float ang = 6.283185307179586f * (float)(mm * n) * (1.0f / 512.0f);
    float c = (mm == 0) ? (1.0f / 512.0f) : (2.0f / 512.0f);
    float v = (j < 64) ? c * cosf(ang) : ((j == 64) ? 0.0f : -c * sinf(ang));
    idftT[(size_t)n * 128 + j] = (__bf16)v;
  }
}

// ---------------- per-row layernorm (mean/var over D), one wave32 per row
__global__ void k_ln_row(const float* __restrict__ x, const float* __restrict__ g,
                         const float* __restrict__ bb, float* __restrict__ out) {
  int gid = blockIdx.x * blockDim.x + threadIdx.x;
  int row = gid >> 5, lane = gid & 31;
  if (row >= MROWS) return;
  const float* xp = x + (size_t)row * Dc;
  float s = 0.f, s2 = 0.f;
  for (int i = lane; i < Dc; i += 32) { float v = xp[i]; s += v; s2 += v * v; }
#pragma unroll
  for (int off = 16; off > 0; off >>= 1) {
    s  += __shfl_xor(s, off, 32);
    s2 += __shfl_xor(s2, off, 32);
  }
  float mu = s * (1.0f / Dc);
  float var = s2 * (1.0f / Dc) - mu * mu;
  float rs = rsqrtf(var + 1e-5f);
  float* op = out + (size_t)row * Dc;
  for (int i = lane; i < Dc; i += 32) op[i] = (xp[i] - mu) * rs * g[i] + bb[i];
}

// subtract mean over token axis (my_Layernorm tail), emit bf16 for the head GEMM
__global__ void k_ln_sub(const float* __restrict__ xh, __bf16* __restrict__ outB) {
  int t = blockIdx.x * blockDim.x + threadIdx.x;      // (b,d)
  if (t >= Bc * Dc) return;
  size_t base = (size_t)(t >> 9) * Nc * Dc + (t & (Dc - 1));
  const float* xp = xh + base;
  float s = 0.f;
  for (int n = 0; n < Nc; ++n) s += xp[(size_t)n * Dc];
  float mu = s * (1.0f / Nc);
  __bf16* op = outB + base;
  for (int n = 0; n < Nc; ++n) op[(size_t)n * Dc] = (__bf16)(xp[(size_t)n * Dc] - mu);
}

// ---------------- per-batch 512x512 bf16 transpose (q -> qT for the DFT GEMM)
__global__ void k_transpose_bf16(const __bf16* __restrict__ src, __bf16* __restrict__ dst,
                                 int R, int C) {
  __shared__ __bf16 tile[32][33];
  const __bf16* sp = src + (size_t)blockIdx.z * R * C;
  __bf16* dp = dst + (size_t)blockIdx.z * R * C;
  int c0 = blockIdx.x * 32, r0 = blockIdx.y * 32;
  for (int i = threadIdx.y; i < 32; i += 8)
    tile[i][threadIdx.x] = sp[(size_t)(r0 + i) * C + c0 + threadIdx.x];
  __syncthreads();
  for (int i = threadIdx.y; i < 32; i += 8)
    dp[(size_t)(c0 + i) * R + r0 + threadIdx.x] = tile[threadIdx.x][i];
}

// ---------------- weight conversion f32 -> bf16 (plain / transposed)
__global__ void k_cvt(const float* __restrict__ s, __bf16* __restrict__ d, int n) {
  int i = blockIdx.x * blockDim.x + threadIdx.x;
  if (i < n) d[i] = (__bf16)s[i];
}
__global__ void k_cvt_t(const float* __restrict__ s, __bf16* __restrict__ d, int R, int C) {
  int i = blockIdx.x * blockDim.x + threadIdx.x;
  if (i < R * C) { int r = i / C, c = i % C; d[(size_t)c * R + r] = (__bf16)s[i]; }
}

// ---------------- final: d_out[b,p,n] = Sum[b,n,p]
__global__ void k_final(const float* __restrict__ S, float* __restrict__ out) {
  int idx = blockIdx.x * blockDim.x + threadIdx.x;
  if (idx >= Bc * PREDc * Nc) return;
  int n = idx & 511;
  int p = (idx >> 9) % PREDc;
  int b = idx / (PREDc * Nc);
  out[idx] = S[((size_t)b * Nc + n) * PREDc + p];
}

extern "C" void kernel_launch(void* const* d_in, const int* in_sizes, int n_in,
                              void* d_out, int out_size, void* d_ws, size_t ws_size,
                              hipStream_t stream) {
  (void)in_sizes; (void)n_in; (void)out_size; (void)ws_size;
  const float* x_enc   = (const float*)d_in[0];
  const float* W_trend = (const float*)d_in[4];
  const float* b_trend = (const float*)d_in[5];
  const float* W_emb   = (const float*)d_in[6];
  const float* b_emb   = (const float*)d_in[7];
  const float* Wq      = (const float*)d_in[8];
  const float* bq      = (const float*)d_in[9];
  // d_in[10..13] = Wk,bk,Wv,bv: computed-but-unused in the reference -> skipped.
  const float* fw_r    = (const float*)d_in[14];
  const float* fw_i    = (const float*)d_in[15];
  const float* Wo      = (const float*)d_in[16];
  const float* bo      = (const float*)d_in[17];
  const float* W1      = (const float*)d_in[18];
  const float* W2      = (const float*)d_in[19];
  const float* ln_g    = (const float*)d_in[20];
  const float* ln_b    = (const float*)d_in[21];
  const float* W_head  = (const float*)d_in[22];
  const float* b_head  = (const float*)d_in[23];
  float* out = (float*)d_out;

  char* ws = (char*)d_ws;
  size_t off = 0;
  auto alloc = [&](size_t bytes) {
    size_t o = off; off += (bytes + 255) & ~(size_t)255; return o;
  };
  const size_t actF = (size_t)MROWS * Dc * 4;   // 32 MB
  const size_t actB = (size_t)MROWS * Dc * 2;   // 16 MB

  // bf16 weight pool (pre-transposed for the GEMM B operand)
  __bf16* wembT  = (__bf16*)(ws + alloc(512 * 512 * 2));
  __bf16* wqT[2], *woT[2], *w1b[2], *w2b[2];
  for (int il = 0; il < 2; ++il) {
    wqT[il] = (__bf16*)(ws + alloc(512 * 512 * 2));
    woT[il] = (__bf16*)(ws + alloc(512 * 512 * 2));
    w1b[il] = (__bf16*)(ws + alloc((size_t)DFc * Dc * 2));   // x@W1.T: Bt = W1 as-is
    w2b[il] = (__bf16*)(ws + alloc((size_t)DFc * Dc * 2));   // y@W2.T: Bt = W2 as-is
  }
  __bf16* wheadT = (__bf16*)(ws + alloc((size_t)PREDc * 512 * 2));
  __bf16* wtT    = (__bf16*)(ws + alloc((size_t)PREDc * 512 * 2));
  __bf16* dftB   = (__bf16*)(ws + alloc(128 * 512 * 2));
  __bf16* idftB  = (__bf16*)(ws + alloc(512 * 128 * 2));

  // activations
  __bf16* trendT = (__bf16*)(ws + alloc(actB));  // live until the trend head GEMM
  float*  encF   = (float*) (ws + alloc(actF));
  __bf16* encB   = (__bf16*)(ws + alloc(actB));
  __bf16* bufB1  = (__bf16*)(ws + alloc((size_t)MROWS * DFc * 2)); // q / xt / h1
  __bf16* bufB2  = (__bf16*)(ws + alloc(actB));  // seasonalT, then qT
  float*  specF  = (float*) (ws + alloc((size_t)MROWS * 128 * 4)); // rfft modes
  __bf16* specB  = (__bf16*)(ws + alloc((size_t)MROWS * 128 * 2)); // mixed modes
  float*  bufF1  = (float*) (ws + alloc(actF));  // attn-res / ffn-res / ln tmp
  float*  xF     = (float*) (ws + alloc(actF));
  __bf16* xB     = (__bf16*)(ws + alloc(actB));
  float*  headS  = (float*) (ws + alloc((size_t)MROWS * PREDc * 4));
  float*  headU  = (float*) (ws + alloc((size_t)MROWS * PREDc * 4));
  __bf16* seasT  = bufB2;
  __bf16* qT     = bufB2;

  const dim3 thr(256);
  const dim3 g512(4, 128);    // N=512:  4 x (M/128)
  const dim3 g2048(16, 128);  // N=2048
  const dim3 g128(1, 128);    // N=128 (DFT modes)
  const dim3 g96(1, 128);     // N=96 (heads; TDM zero-fills rows 96..127)

  // 0) DFT bases + bf16 weights
  k_fill_basis<<<256, thr, 0, stream>>>(dftB, idftB);
  k_cvt_t<<<1024, thr, 0, stream>>>(W_emb, wembT, 512, 512);
  for (int il = 0; il < 2; ++il) {
    k_cvt_t<<<1024, thr, 0, stream>>>(Wq + (size_t)il * 512 * 512, wqT[il], 512, 512);
    k_cvt_t<<<1024, thr, 0, stream>>>(Wo + (size_t)il * 512 * 512, woT[il], 512, 512);
    k_cvt<<<4096, thr, 0, stream>>>(W1 + (size_t)il * DFc * Dc, w1b[il], DFc * Dc);
    k_cvt<<<4096, thr, 0, stream>>>(W2 + (size_t)il * DFc * Dc, w2b[il], DFc * Dc);
  }
  k_cvt_t<<<192, thr, 0, stream>>>(W_head,  wheadT, 512, PREDc);
  k_cvt_t<<<192, thr, 0, stream>>>(W_trend, wtT,    512, PREDc);

  // 1) input decomposition -> seasonalT/trendT [B,N,L] bf16 (transposed = GEMM A)
  k_decomp_in<<<64, thr, 0, stream>>>(x_enc, seasT, trendT);

  // 2) variate-token embedding: enc[b,n,:] = seasonalT[b,n,:] @ W_emb + b_emb
  k_gemm_bf16<<<g512, thr, 0, stream>>>(seasT, wembT, MROWS, 512, 512,
                                        b_emb, nullptr, 0, encF, encB);

  // 3) encoder layers
  for (int il = 0; il < 2; ++il) {
    // q projection (k/v are dead code in the reference)
    k_gemm_bf16<<<g512, thr, 0, stream>>>(encB, wqT[il], MROWS, 512, 512,
                                          bq + (size_t)il * 512, nullptr, 0,
                                          nullptr, bufB1);
    // q[B,N,D] -> qT[B,D,N] so the token axis is the GEMM K axis
    k_transpose_bf16<<<dim3(16, 16, Bc), dim3(32, 8), 0, stream>>>(bufB1, qT, 512, 512);
    // rfft (64 low modes) as a GEMM against the cos/sin basis
    k_gemm_bf16<<<g128, thr, 0, stream>>>(qT, dftB, MROWS, 128, 512,
                                          nullptr, nullptr, 0, specF, nullptr);
    // complex per-(h,m) 64x64 mode mixing
    k_mode_mix<<<512, thr, 0, stream>>>(specF, fw_r + (size_t)il * Hc * Ec * Ec * MODESc,
                                        fw_i + (size_t)il * Hc * Ec * Ec * MODESc, specB);
    // irfft as GEMM; output rows (h*64+o) == the reference's scrambled reshape rows
    k_gemm_bf16<<<g512, thr, 0, stream>>>(specB, idftB, MROWS, 512, 128,
                                          nullptr, nullptr, 0, nullptr, bufB1);
    // Wo projection + residual
    k_gemm_bf16<<<g512, thr, 0, stream>>>(bufB1, woT[il], MROWS, 512, 512,
                                          bo + (size_t)il * 512, encF, 0, bufF1, nullptr);
    // series_decomp over tokens
    k_decomp_tok<<<64, thr, 0, stream>>>(bufF1, xF, xB);
    // FFN: gelu(x@W1.T) @ W2.T + residual
    k_gemm_bf16<<<g2048, thr, 0, stream>>>(xB, w1b[il], MROWS, DFc, 512,
                                           nullptr, nullptr, 1, nullptr, bufB1);
    k_gemm_bf16<<<g512, thr, 0, stream>>>(bufB1, w2b[il], MROWS, 512, DFc,
                                          nullptr, xF, 0, bufF1, nullptr);
    k_decomp_tok<<<64, thr, 0, stream>>>(bufF1, encF, encB);
  }

  // 4) my_layernorm
  k_ln_row<<<2048, thr, 0, stream>>>(encF, ln_g, ln_b, bufF1);
  k_ln_sub<<<64, thr, 0, stream>>>(bufF1, xB);

  // 5) seasonal head + trend head (fused via residual), then transpose to [B,P,N]
  k_gemm_bf16<<<g96, thr, 0, stream>>>(xB, wheadT, MROWS, PREDc, 512,
                                       b_head, nullptr, 0, headS, nullptr);
  k_gemm_bf16<<<g96, thr, 0, stream>>>(trendT, wtT, MROWS, PREDc, 512,
                                       b_trend, headS, 0, headU, nullptr);
  k_final<<<(Bc * PREDc * Nc) / 256, thr, 0, stream>>>(headU, out);
}
